// A_ELADMMNet_42520176230987
// MI455X (gfx1250) — matrix operand order repaired and checked by reading, not compiled, not run on hardware
//
#include <hip/hip_runtime.h>
#include <hip/hip_bf16.h>

// ---------------------------------------------------------------------------
// CDNA5 (gfx1250) wave32 WMMA types & helpers
// ---------------------------------------------------------------------------
typedef __attribute__((ext_vector_type(16))) __bf16 bf16x16;
typedef __attribute__((ext_vector_type(8)))  __bf16 bf16x8;
typedef __attribute__((ext_vector_type(8)))  float  f32x8;

__device__ __forceinline__ bf16x8 bzero8() {
    bf16x8 z;
#pragma unroll
    for (int i = 0; i < 8; ++i) z[i] = (__bf16)0.0f;
    return z;
}
__device__ __forceinline__ f32x8 fzero8() {
    f32x8 z;
#pragma unroll
    for (int i = 0; i < 8; ++i) z[i] = 0.0f;
    return z;
}
__device__ __forceinline__ bf16x16 bcat(bf16x8 lo, bf16x8 hi) {
    return __builtin_shufflevector(lo, hi, 0,1,2,3,4,5,6,7,8,9,10,11,12,13,14,15);
}
__device__ __forceinline__ f32x8 wmma_bf16(bf16x16 a, bf16x16 b, f32x8 c) {
    // v_wmma_f32_16x16x32_bf16  D = A*B + C   (wave32, EXEC all ones)
    return __builtin_amdgcn_wmma_f32_16x16x32_bf16(false, a, false, b, (short)0, c, false, false);
}

// Async global->LDS DMA (CDNA5): per-lane 16B copy, tracked by ASYNCcnt.
// %0 = LDS byte address (low 32 bits of generic shared pointer), %1 = global addr.
__device__ __forceinline__ void async_copy_b128(unsigned lds_addr, const void* gaddr) {
    asm volatile("global_load_async_to_lds_b128 %0, %1, off"
                 :: "v"(lds_addr), "v"(gaddr)
                 : "memory");
}
__device__ __forceinline__ unsigned lds_off(const void* p) {
    return (unsigned)(unsigned long long)p;   // low 32 bits of shared aperture addr
}

// Problem constants
#define BB   4096
#define MM   327
#define NN   800
#define LAY  9
#define IMH  20
#define IMW  40
#define BN   (4096L * 800L)     // 3,276,800 state elements / pixels
#define KP   352                // 327 padded to x32

// ---------------------------------------------------------------------------
// Generic bf16 TN GEMM with f32 accumulation:
//   C[M x N] = A[M x K] * Bt[N x K]^T     (Bt stores B column-major)
// Block = 256 thr = 8 waves; tile 128(M) x 64(N), K-step 32.
// Double-buffered LDS filled by global_load_async_to_lds_b128; OOB rows are
// clamped (their products only land in OOB C rows/cols, which are never
// stored), so control flow is fully uniform.
// ---------------------------------------------------------------------------
__global__ __launch_bounds__(256)
void k_gemm_tn_bf16(const __bf16* __restrict__ A, const __bf16* __restrict__ Bt,
                    float* __restrict__ C, int M, int N, int K,
                    int lda, int ldb, int ldc)
{
    __shared__ __bf16 As[2][128 * 40];   // row stride 40 bf16 (80B): 16B-aligned chunks
    __shared__ __bf16 Bs[2][64  * 40];

    const int tid  = threadIdx.x;
    const int lane = tid & 31;
    const int wave = tid >> 5;
    const int m0 = blockIdx.y * 128;
    const int n0 = blockIdx.x * 64;
    const int wm = (wave & 3) * 32;     // wave M offset in tile
    const int wn = (wave >> 2) * 32;    // wave N offset in tile
    const int hrow = lane & 15;
    const int hk   = lane >> 4;

    // ---- per-thread staging assignment (fixed across K loop)
    // A tile: 128 rows x 4 chunks = 512 chunks -> 2 per thread
    // B tile:  64 rows x 4 chunks = 256 chunks -> 1 per thread
    const int rA0 = tid >> 2;                 // 0..63
    const int rA1 = (tid + 256) >> 2;         // 64..127
    const int q   = tid & 3;                  // 16B chunk within 32-K row
    const int grA0 = min(m0 + rA0, M - 1);    // clamped: harmless for stores
    const int grA1 = min(m0 + rA1, M - 1);
    const int grB  = min(n0 + rA0, N - 1);
    const __bf16* gA0 = A  + (size_t)grA0 * lda + q * 8;
    const __bf16* gA1 = A  + (size_t)grA1 * lda + q * 8;
    const __bf16* gB  = Bt + (size_t)grB  * ldb + q * 8;
    unsigned lA0[2], lA1[2], lB[2];
#pragma unroll
    for (int s = 0; s < 2; ++s) {
        lA0[s] = lds_off(&As[s][rA0 * 40 + q * 8]);
        lA1[s] = lds_off(&As[s][rA1 * 40 + q * 8]);
        lB[s]  = lds_off(&Bs[s][rA0 * 40 + q * 8]);
    }

    f32x8 acc[2][2];
#pragma unroll
    for (int a = 0; a < 2; ++a)
#pragma unroll
        for (int b = 0; b < 2; ++b) acc[a][b] = fzero8();

    // prologue: stage tile 0 into buffer 0
    async_copy_b128(lA0[0], gA0);
    async_copy_b128(lA1[0], gA1);
    async_copy_b128(lB[0],  gB);

    const int nk = K >> 5;
    for (int j = 0; j < nk; ++j) {
        const int cur = j & 1;
        if (j + 1 < nk) {
            // prefetch next tile into the other buffer, then wait for current
            const int koff = (j + 1) * 32;
            async_copy_b128(lA0[cur ^ 1], gA0 + koff);
            async_copy_b128(lA1[cur ^ 1], gA1 + koff);
            async_copy_b128(lB[cur ^ 1],  gB  + koff);
            asm volatile("s_wait_asynccnt 0x3" ::: "memory");  // 3 outstanding = next tile only
        } else {
            asm volatile("s_wait_asynccnt 0x0" ::: "memory");
        }
        __syncthreads();

        // ---- fragments per documented 16-bit layouts
        bf16x16 afr[2], bfr[2];
#pragma unroll
        for (int mf = 0; mf < 2; ++mf) {
            const __bf16* p = &As[cur][(wm + mf * 16 + hrow) * 40 + hk * 8];
            afr[mf] = bcat(*reinterpret_cast<const bf16x8*>(p),
                           *reinterpret_cast<const bf16x8*>(p + 16));
        }
#pragma unroll
        for (int nf = 0; nf < 2; ++nf) {
            const __bf16* p = &Bs[cur][(wn + nf * 16 + hrow) * 40 + hk * 8];
            bfr[nf] = bcat(*reinterpret_cast<const bf16x8*>(p),
                           *reinterpret_cast<const bf16x8*>(p + 16));
        }
#pragma unroll
        for (int mf = 0; mf < 2; ++mf)
#pragma unroll
            for (int nf = 0; nf < 2; ++nf)
                acc[mf][nf] = wmma_bf16(afr[mf], bfr[nf], acc[mf][nf]);
        __syncthreads();   // all waves done reading before this buffer is refilled
    }

    // ---- store: D row = r + hk*8 (+16*mf), col = lane&15 (+16*nf)
#pragma unroll
    for (int mf = 0; mf < 2; ++mf)
#pragma unroll
        for (int nf = 0; nf < 2; ++nf)
#pragma unroll
            for (int r = 0; r < 8; ++r) {
                const int row = m0 + wm + mf * 16 + hk * 8 + r;
                const int col = n0 + wn + nf * 16 + hrow;
                if (row < M && col < N)
                    C[(size_t)row * ldc + col] = acc[mf][nf][r];
            }
}

// ---------------------------------------------------------------------------
// 3x3 SAME conv, 32->32 channels, NHWC bf16, implicit-GEMM on WMMA.
// K = 9 taps x 32 channels.  Block = 256 thr = 8 waves x 16 pixels.
// Wmat[o][t*32+c] staged in LDS.
// ---------------------------------------------------------------------------
__global__ __launch_bounds__(256)
void k_conv3x3_wmma(const __bf16* __restrict__ in,   // [B*800][32]
                    const __bf16* __restrict__ Wmat, // [32][288]
                    __bf16* __restrict__ out, int relu)
{
    __shared__ __bf16 Wl[32 * 288];
    const int tid = threadIdx.x;
    for (int i = tid; i < 32 * 288; i += 256) Wl[i] = Wmat[i];
    __syncthreads();

    const int lane = tid & 31;
    const int wave = tid >> 5;
    const int hk   = lane >> 4;
    const int p    = blockIdx.x * 128 + wave * 16 + (lane & 15);
    const int b    = p / 800;
    const int rem  = p % 800;
    const int y    = rem / IMW;
    const int x    = rem % IMW;

    f32x8 acc[2];
    acc[0] = fzero8();
    acc[1] = fzero8();

#pragma unroll
    for (int t = 0; t < 9; ++t) {
        const int yy = y + t / 3 - 1;
        const int xx = x + t % 3 - 1;
        bf16x8 blo = bzero8(), bhi = bzero8();
        if (yy >= 0 && yy < IMH && xx >= 0 && xx < IMW) {
            const __bf16* src = in + ((size_t)b * 800 + yy * IMW + xx) * 32 + hk * 8;
            blo = *reinterpret_cast<const bf16x8*>(src);
            bhi = *reinterpret_cast<const bf16x8*>(src + 16);
        }
        const bf16x16 bfr = bcat(blo, bhi);
#pragma unroll
        for (int mf = 0; mf < 2; ++mf) {
            const __bf16* wp = &Wl[(mf * 16 + (lane & 15)) * 288 + t * 32 + hk * 8];
            const bf16x16 afr = bcat(*reinterpret_cast<const bf16x8*>(wp),
                                     *reinterpret_cast<const bf16x8*>(wp + 16));
            acc[mf] = wmma_bf16(afr, bfr, acc[mf]);
        }
    }

#pragma unroll
    for (int mf = 0; mf < 2; ++mf)
#pragma unroll
        for (int r = 0; r < 8; ++r) {
            float v = acc[mf][r];
            if (relu) v = fmaxf(v, 0.0f);
            const int o = mf * 16 + hk * 8 + r;
            out[(size_t)p * 32 + o] = (__bf16)v;
        }
}

// ---------------------------------------------------------------------------
// Small convs + elementwise kernels (scalar f32)
// ---------------------------------------------------------------------------
__global__ __launch_bounds__(256)
void k_conv1f(const float* __restrict__ z, const float* __restrict__ w, // w[32][1][3][3]
              __bf16* __restrict__ out)
{
    __shared__ float Wl[288];
    const int tid = threadIdx.x;
    for (int i = tid; i < 288; i += 256) Wl[i] = w[i];
    __syncthreads();
    const long p = (long)blockIdx.x * 256 + tid;
    const int b = (int)(p / 800), rem = (int)(p % 800);
    const int y = rem / IMW, x = rem % IMW;
    float tap[9];
#pragma unroll
    for (int t = 0; t < 9; ++t) {
        const int yy = y + t / 3 - 1, xx = x + t % 3 - 1;
        tap[t] = (yy >= 0 && yy < IMH && xx >= 0 && xx < IMW)
                 ? z[(size_t)b * 800 + yy * IMW + xx] : 0.0f;
    }
#pragma unroll
    for (int o = 0; o < 32; ++o) {
        float a = 0.0f;
#pragma unroll
        for (int t = 0; t < 9; ++t) a += tap[t] * Wl[o * 9 + t];
        out[(size_t)p * 32 + o] = (__bf16)fmaxf(a, 0.0f);
    }
}

__global__ __launch_bounds__(256)
void k_conv2b(const __bf16* __restrict__ in, const float* __restrict__ w, // w[1][32][3][3]
              const float* __restrict__ sub, float* __restrict__ out)
{
    __shared__ float Wl[288]; // [t*32+c]
    const int tid = threadIdx.x;
    for (int j = tid; j < 288; j += 256) {
        const int t = j >> 5, c = j & 31;
        Wl[j] = w[c * 9 + t];
    }
    __syncthreads();
    const long p = (long)blockIdx.x * 256 + tid;
    const int b = (int)(p / 800), rem = (int)(p % 800);
    const int y = rem / IMW, x = rem % IMW;
    float a = 0.0f;
#pragma unroll
    for (int t = 0; t < 9; ++t) {
        const int yy = y + t / 3 - 1, xx = x + t % 3 - 1;
        if (yy >= 0 && yy < IMH && xx >= 0 && xx < IMW) {
            const __bf16* s = in + ((size_t)b * 800 + yy * IMW + xx) * 32;
#pragma unroll
            for (int c = 0; c < 32; ++c) a += (float)s[c] * Wl[t * 32 + c];
        }
    }
    if (sub) a -= sub[p];
    out[p] = a;
}

__global__ void k_soft(const __bf16* in, __bf16* out, const float* thr, int layer, long n) {
    const long i = (long)blockIdx.x * 256 + threadIdx.x;
    if (i >= n) return;
    const float s = fabsf(thr[layer]);
    const float v = (float)in[i];
    const float m = fmaxf(fabsf(v) - s, 0.0f);
    out[i] = (__bf16)(v < 0.0f ? -m : m);
}

__global__ void k_zero(float* p, long n) {
    const long i = (long)blockIdx.x * 256 + threadIdx.x;
    if (i < n) p[i] = 0.0f;
}

__global__ void k_hatx(const float* X, const float* Xp, const float* th, int layer, int useT,
                       float* hx, __bf16* hxb, long n) {
    const long i = (long)blockIdx.x * 256 + threadIdx.x;
    if (i >= n) return;
    const float t = useT ? th[layer] : 0.0f;
    const float xv = X[i];
    const float h = xv + t * (xv - Xp[i]);
    hx[i] = h;
    hxb[i] = (__bf16)h;
}

__global__ void k_xnew(const float* hx, const float* PhiTb, const float* GX,
                       const float* Zs, const float* Ls,
                       const float* hArr, const float* b1Arr, int layer,
                       float* Xn, long n) {
    const long i = (long)blockIdx.x * 256 + threadIdx.x;
    if (i >= n) return;
    const float h = hArr[layer], b1 = b1Arr[layer];
    const float hv = hx[i];
    Xn[i] = hv + h * (PhiTb[i] - GX[i] + b1 * (Zs[i] - hv) - Ls[i]);
}

__global__ void k_zflat(const float* Z, const float* Zp, const float* Ls, const float* Xn,
                        const float* tzArr, const float* hArr, const float* b2Arr,
                        int layer, int useT, float* zf, long n) {
    const long i = (long)blockIdx.x * 256 + threadIdx.x;
    if (i >= n) return;
    const float tz = useT ? tzArr[layer] : 0.0f;
    const float h = hArr[layer], b2 = b2Arr[layer];
    const float zv = Z[i];
    const float hz = zv + tz * (zv - Zp[i]);
    zf[i] = hz + h * (Ls[i] + b2 * (Xn[i] - hz));
}

__global__ void k_lnew(const float* L, const float* Lp, const float* Xn, const float* Zn,
                       const float* tlArr, const float* hArr, const float* b1Arr,
                       int layer, int useT, float* Ln, long n) {
    const long i = (long)blockIdx.x * 256 + threadIdx.x;
    if (i >= n) return;
    const float tl = useT ? tlArr[layer] : 0.0f;
    const float h = hArr[layer], b1 = b1Arr[layer];
    const float lv = L[i];
    const float hl = lv + tl * (lv - Lp[i]);
    Ln[i] = hl + h * b1 * (Xn[i] - Zn[i]);
}

// ---- conversions / packing ----
__global__ void k_f2b(const float* in, __bf16* out, long n) {
    const long i = (long)blockIdx.x * 256 + threadIdx.x;
    if (i < n) out[i] = (__bf16)in[i];
}
// out[R][Kp] = pad(in[R][K])
__global__ void k_pad_b(const float* in, __bf16* out, int R, int K, int Kp) {
    const long i = (long)blockIdx.x * 256 + threadIdx.x;
    if (i >= (long)R * Kp) return;
    const int r = (int)(i / Kp), c = (int)(i % Kp);
    out[i] = (c < K) ? (__bf16)in[(size_t)r * K + c] : (__bf16)0.0f;
}
// out[C][Kp] = pad(in^T), in is [R][C]
__global__ void k_tr_pad_b(const float* in, __bf16* out, int R, int C, int Kp) {
    const long i = (long)blockIdx.x * 256 + threadIdx.x;
    if (i >= (long)C * Kp) return;
    const int r = (int)(i / Kp), k = (int)(i % Kp);
    out[i] = (k < R) ? (__bf16)in[(size_t)k * C + r] : (__bf16)0.0f;
}
// conv weight [O=32][C=32][3][3] -> Wmat[o][t*32+c]
__global__ void k_prep_wmat(const float* w, __bf16* out) {
    const int i = blockIdx.x * 256 + threadIdx.x;
    if (i >= 32 * 288) return;
    const int o = i / 288, r = i % 288;
    const int t = r >> 5, c = r & 31;
    out[i] = (__bf16)w[((size_t)o * 32 + c) * 9 + t];
}
__global__ void k_wloss(const float* t, float* out) {
    const int i = blockIdx.x * 256 + threadIdx.x;
    if (i >= MM * MM) return;
    const int r = i / MM, c = i % MM;
    out[i] = t[i] - (r == c ? 1.0f : 0.0f);
}

// ---------------------------------------------------------------------------
// Launcher
// ---------------------------------------------------------------------------
extern "C" void kernel_launch(void* const* d_in, const int* in_sizes, int n_in,
                              void* d_out, int out_size, void* d_ws, size_t ws_size,
                              hipStream_t stream) {
    (void)in_sizes; (void)n_in; (void)out_size; (void)ws_size;

    const float* y    = (const float*)d_in[0];
    const float* W    = (const float*)d_in[2];
    const float* b1   = (const float*)d_in[3];
    const float* b2   = (const float*)d_in[4];
    const float* hArr = (const float*)d_in[5];
    const float* sthr = (const float*)d_in[6];
    const float* thx  = (const float*)d_in[7];
    const float* thz  = (const float*)d_in[8];
    const float* thL  = (const float*)d_in[9];
    const float* c1f  = (const float*)d_in[10];
    const float* c2f  = (const float*)d_in[11];
    const float* c1b  = (const float*)d_in[12];
    const float* c2b  = (const float*)d_in[13];

    float* outZ = (float*)d_out;
    float* outS = outZ + (size_t)LAY * BN;
    float* outW = outS + (size_t)LAY * BN;

    // carve workspace
    char* w = (char*)d_ws;
    auto alloc = [&](size_t bytes) -> void* {
        void* p = (void*)w;
        w += (bytes + 255) & ~(size_t)255;
        return p;
    };
    float*  zeros   = (float*)alloc(BN * 4);
    float*  xbuf[3] = { (float*)alloc(BN * 4), (float*)alloc(BN * 4), (float*)alloc(BN * 4) };
    float*  lbuf[3] = { (float*)alloc(BN * 4), (float*)alloc(BN * 4), (float*)alloc(BN * 4) };
    float*  PhiTb   = (float*)alloc(BN * 4);
    float*  GX      = (float*)alloc(BN * 4);
    float*  hatxF   = (float*)alloc(BN * 4);
    float*  zflat   = (float*)alloc(BN * 4);
    __bf16* hatxB   = (__bf16*)alloc(BN * 2);
    float*  ppF     = (float*)alloc((size_t)NN * NN * 4);
    __bf16* ppB     = (__bf16*)alloc((size_t)NN * NN * 2);
    __bf16* WtB     = (__bf16*)alloc((size_t)NN * KP * 2);
    __bf16* yB      = (__bf16*)alloc((size_t)BB * KP * 2);
    __bf16* WB      = (__bf16*)alloc((size_t)MM * NN * 2);
    float*  wlT     = (float*)alloc((size_t)MM * MM * 4);
    __bf16* WmatF   = (__bf16*)alloc(32 * 288 * 2);
    __bf16* WmatB   = (__bf16*)alloc(32 * 288 * 2);
    __bf16* actA    = (__bf16*)alloc(BN * 32 * 2);
    __bf16* actB    = (__bf16*)alloc(BN * 32 * 2);
    __bf16* actC    = (__bf16*)alloc(BN * 32 * 2);

    const int NB   = (int)(BN / 256);          // 12800 (exact)
    const long NCH = BN * 32;                   // 104,857,600
    const int NBC  = (int)((NCH + 255) / 256);  // 409,600
    auto gb = [](long n) { return (int)((n + 255) / 256); };

    // ---- one-time prep ----
    k_zero<<<NB, 256, 0, stream>>>(zeros, BN);
    k_tr_pad_b<<<gb((long)NN * KP), 256, 0, stream>>>(W, WtB, MM, NN, KP);
    k_pad_b  <<<gb((long)BB * KP), 256, 0, stream>>>(y, yB, BB, MM, KP);
    k_f2b    <<<gb((long)MM * NN), 256, 0, stream>>>(W, WB, (long)MM * NN);
    k_prep_wmat<<<36, 256, 0, stream>>>(c2f, WmatF);
    k_prep_wmat<<<36, 256, 0, stream>>>(c1b, WmatB);

    // PhiTPhi = W^T W    [800x800], K=327 (padded 352)
    k_gemm_tn_bf16<<<dim3((NN + 63) / 64, (NN + 127) / 128), 256, 0, stream>>>(
        WtB, WtB, ppF, NN, NN, KP, KP, KP, NN);
    k_f2b<<<gb((long)NN * NN), 256, 0, stream>>>(ppF, ppB, (long)NN * NN);
    // PhiTb = y @ W      [4096x800]
    k_gemm_tn_bf16<<<dim3((NN + 63) / 64, (BB + 127) / 128), 256, 0, stream>>>(
        yB, WtB, PhiTb, BB, NN, KP, KP, KP, NN);
    // Wloss = W W^T - I  [327x327]
    k_gemm_tn_bf16<<<dim3((MM + 63) / 64, (MM + 127) / 128), 256, 0, stream>>>(
        WB, WB, wlT, MM, MM, NN, NN, NN, MM);
    k_wloss<<<gb((long)MM * MM), 256, 0, stream>>>(wlT, outW);

    // ---- layer loop ----
    const float* xCur = zeros; const float* xPrev = zeros;
    const float* lCur = zeros; const float* lPrev = zeros;

    for (int i = 0; i < LAY; ++i) {
        const int useT = (i >= 1) ? 1 : 0;
        const float* ZselX = (i <= 1) ? zeros : (outZ + (size_t)(i - 1) * BN);
        const float* LselX = (i <= 1) ? zeros : lCur;
        const float* Zcur  = (i == 0) ? zeros : (outZ + (size_t)(i - 1) * BN);
        const float* Zprv  = (i <= 1) ? zeros : (outZ + (size_t)(i - 2) * BN);
        const float* LselZ = (i <= 1) ? zeros : lCur;
        float* Xn = xbuf[i % 3];
        float* Ln = lbuf[i % 3];

        k_hatx<<<NB, 256, 0, stream>>>(xCur, xPrev, thx, i, useT, hatxF, hatxB, BN);
        k_gemm_tn_bf16<<<dim3((NN + 63) / 64, (BB + 127) / 128), 256, 0, stream>>>(
            hatxB, ppB, GX, BB, NN, NN, NN, NN, NN);
        k_xnew<<<NB, 256, 0, stream>>>(hatxF, PhiTb, GX, ZselX, LselX, hArr, b1, i, Xn, BN);
        k_zflat<<<NB, 256, 0, stream>>>(Zcur, Zprv, LselZ, Xn, thz, hArr, b2, i, useT, zflat, BN);

        // conv block
        k_conv1f<<<NB, 256, 0, stream>>>(zflat, c1f, actA);                 // relu(conv1f(z))
        k_conv3x3_wmma<<<(int)(BN / 128), 256, 0, stream>>>(actA, WmatF, actB, 0);  // x_fwd
        k_soft<<<NBC, 256, 0, stream>>>(actB, actC, sthr, i, NCH);          // soft(x_fwd)
        k_conv3x3_wmma<<<(int)(BN / 128), 256, 0, stream>>>(actC, WmatB, actA, 1);  // relu(conv1b(soft))
        k_conv2b<<<NB, 256, 0, stream>>>(actA, c2b, nullptr, outZ + (size_t)i * BN); // x_back -> Z_i
        k_conv3x3_wmma<<<(int)(BN / 128), 256, 0, stream>>>(actB, WmatB, actA, 1);  // relu(conv1b(x_fwd))
        k_conv2b<<<NB, 256, 0, stream>>>(actA, c2b, zflat, outS + (size_t)i * BN);   // x_sym - z_in

        k_lnew<<<NB, 256, 0, stream>>>(lCur, lPrev, Xn, outZ + (size_t)i * BN,
                                       thL, hArr, b1, i, useT, Ln, BN);

        xPrev = xCur; xCur = Xn;
        lPrev = lCur; lCur = Ln;
    }
}
